// CrossAttentionFusion_12051678233322
// MI455X (gfx1250) — compile-verified, matrix operand order checked by reading, not run on hardware
//
#include <hip/hip_runtime.h>
#include <hip/hip_bf16.h>
#include <math.h>

typedef __attribute__((ext_vector_type(16))) __bf16 v16bf;
typedef __attribute__((ext_vector_type(8)))  __bf16 v8bf;
typedef __attribute__((ext_vector_type(8)))  float  v8f;

#define PIX   16384   // H*W
#define CO    128
#define CHUNK 2048

// Build a 16x32 WMMA A/B frag from two 16-byte-aligned runs of 8 bf16.
__device__ __forceinline__ v16bf frag_bf16(const __bf16* plo, const __bf16* phi) {
    v8bf lo = *(const v8bf*)plo;
    v8bf hi = *(const v8bf*)phi;
    return __builtin_shufflevector(lo, hi, 0, 1, 2, 3, 4, 5, 6, 7,
                                           8, 9, 10, 11, 12, 13, 14, 15);
}

// ---------------------------------------------------------------------------
// Projection: out[o,p] = ReLU( (sum_c w[o,c]*x[c,p] - mean)*scale + beta )
// One wave per 16x16 (o x p) tile.  A = weights (16x32), B = activations.
// WRITE_F32 -> fp32 (C,P) copy; else bf16 (C,P) copy.  Always bf16 (P,C)
// transposed copy scaled by tscale.
// ---------------------------------------------------------------------------
template <bool WRITE_F32>
__global__ void __launch_bounds__(128)
proj_kernel(const float* __restrict__ x,      // [B][Cin][P]
            const float* __restrict__ w,      // [CO][Cin]
            const float* __restrict__ gamma,
            const float* __restrict__ beta,
            const float* __restrict__ mean,
            const float* __restrict__ var,
            float*  __restrict__ proj_f32,    // [B][CO][P]  (WRITE_F32)
            __bf16* __restrict__ proj_bf,     // [B][CO][P]  (!WRITE_F32)
            __bf16* __restrict__ proj_t,      // [B][P][CO]
            int Cin, float tscale)
{
    const int lane = threadIdx.x & 31;
    const int wave = threadIdx.x >> 5;
    const int wid  = blockIdx.x * 4 + wave;       // 0 .. 16383
    const int pT   = wid & 1023;                  // 1024 pixel tiles
    const int t    = wid >> 10;
    const int oT   = t & 7;                       // 8 out-channel tiles
    const int b    = t >> 3;                      // batch

    const int h  = (lane >= 16) ? 1 : 0;
    const int ln = lane & 15;
    const int p0 = pT * 16;
    const int o0 = oT * 16;

    const float* xb = x + (size_t)b * Cin * PIX;

    v8f acc = {};
    for (int c0 = 0; c0 < Cin; c0 += 32) {
        // A-frag: weight row o = o0+ln ; klo = c0+8h+[0..7], khi = +16
        const float* wr = w + (size_t)(o0 + ln) * Cin + c0 + h * 8;
        v16bf af;
        #pragma unroll
        for (int i = 0; i < 8; ++i) {
            af[i]     = (__bf16)wr[i];
            af[8 + i] = (__bf16)wr[16 + i];
        }
        // B-frag: pixel column p0+ln ; K values are channels (stride PIX)
        const float* xc = xb + (size_t)(c0 + h * 8) * PIX + (p0 + ln);
        v16bf bf;
        #pragma unroll
        for (int i = 0; i < 8; ++i) {
            bf[i]     = (__bf16)xc[(size_t)i * PIX];
            bf[8 + i] = (__bf16)xc[(size_t)(16 + i) * PIX];
        }
        acc = __builtin_amdgcn_wmma_f32_16x16x32_bf16(
                  false, af, false, bf, (short)0, acc, false, false);
    }

    // BN + ReLU + stores.  D layout: (o = r + 8h, p = ln)
    #pragma unroll
    for (int r = 0; r < 8; ++r) {
        const int o = o0 + r + 8 * h;
        const int p = p0 + ln;
        const float sc = gamma[o] * rsqrtf(var[o] + 1e-5f);
        float y = (acc[r] - mean[o]) * sc + beta[o];
        y = fmaxf(y, 0.0f);
        const size_t cp_idx = ((size_t)b * CO + o) * PIX + p;
        if (WRITE_F32) proj_f32[cp_idx] = y;
        else           proj_bf[cp_idx]  = (__bf16)y;
        proj_t[((size_t)b * PIX + p) * CO + o] = (__bf16)(y * tscale);
    }
}

// ---------------------------------------------------------------------------
// Flash attention over one 2048-pixel chunk.  One wave per 16-row Q tile.
// Q pre-scaled by 1/sqrt(128).  S = Q K^T; online softmax with the row-sum
// computed as an extra WMMA (P x ones) so the denominator lives in the same
// C/D layout as the O accumulator.  out = w0*img + w1*(O/l).
// ---------------------------------------------------------------------------
__global__ void __launch_bounds__(128)
attn_kernel(const __bf16* __restrict__ q_t,   // [B][P][C]  image (bf16, scaled)
            const __bf16* __restrict__ k_t,   // [B][P][C]  lidar (bf16)
            const __bf16* __restrict__ v_cp,  // [B][C][P]  lidar (bf16)
            const float*  __restrict__ i_cp,  // [B][C][P]  image (f32)
            const float*  __restrict__ mw,    // [2] modality weights
            float* __restrict__ out)          // [B][C][P]
{
    __shared__ alignas(16) __bf16 lds_p[4][16 * 32];   // per-wave P-tile

    const int lane = threadIdx.x & 31;
    const int wave = threadIdx.x >> 5;
    const int h  = lane >> 4;
    const int ln = lane & 15;

    const int wid = blockIdx.x * 4 + wave;    // 0 .. 2047
    const int qT  = wid & 127;                // 128 q tiles / chunk
    const int t   = wid >> 7;
    const int ch  = t & 7;                    // 8 chunks
    const int b   = t >> 3;                   // batch

    const size_t cbase = (size_t)ch * CHUNK;  // pixel base of chunk
    const int    q0    = qT * 16;

    // --- Q A-frags (16 rows x 128 channels -> 4 frags of 16x32) -----------
    const __bf16* qrow = q_t + ((size_t)b * PIX + cbase + q0 + ln) * CO;
    v16bf qf[4];
    #pragma unroll
    for (int kb = 0; kb < 4; ++kb) {
        const __bf16* qp = qrow + kb * 32 + h * 8;
        qf[kb] = frag_bf16(qp, qp + 16);
    }

    // all-ones B frag: row-sum via WMMA (every D column = row sum)
    v16bf ones;
    #pragma unroll
    for (int i = 0; i < 16; ++i) ones[i] = (__bf16)1.0f;

    v8f oacc[8];
    #pragma unroll
    for (int c = 0; c < 8; ++c) oacc[c] = (v8f){};
    v8f lacc = {};                            // softmax denominators
    float m[8];
    #pragma unroll
    for (int r = 0; r < 8; ++r) m[r] = -3.0e38f;

    __bf16* lp = lds_p[wave];

    for (int j0 = 0; j0 < CHUNK; j0 += 32) {
        // ---- S tiles: cols [j0, j0+16) and [j0+16, j0+32) ----------------
        v8f s0 = {}, s1 = {};
        #pragma unroll
        for (int kb = 0; kb < 4; ++kb) {
            const __bf16* kp0 =
                k_t + ((size_t)b * PIX + cbase + j0 + ln) * CO + kb * 32 + h * 8;
            const __bf16* kp1 = kp0 + (size_t)16 * CO;
            __builtin_prefetch((const void*)(kp0 + (size_t)32 * CO), 0, 0);
            v16bf b0 = frag_bf16(kp0, kp0 + 16);
            v16bf b1 = frag_bf16(kp1, kp1 + 16);
            s0 = __builtin_amdgcn_wmma_f32_16x16x32_bf16(
                     false, qf[kb], false, b0, (short)0, s0, false, false);
            s1 = __builtin_amdgcn_wmma_f32_16x16x32_bf16(
                     false, qf[kb], false, b1, (short)0, s1, false, false);
        }

        // ---- online softmax max-update (row = r + 8h spans a 16-lane half)
        float alpha[8];
        #pragma unroll
        for (int r = 0; r < 8; ++r) {
            float v = fmaxf(s0[r], s1[r]);
            #pragma unroll
            for (int off = 1; off < 16; off <<= 1)
                v = fmaxf(v, __shfl_xor(v, off, 32));
            const float mn = fmaxf(m[r], v);
            alpha[r] = __expf(m[r] - mn);
            s0[r] = __expf(s0[r] - mn);
            s1[r] = __expf(s1[r] - mn);
            m[r] = mn;
        }
        #pragma unroll
        for (int r = 0; r < 8; ++r) lacc[r] *= alpha[r];
        #pragma unroll
        for (int c = 0; c < 8; ++c)
            #pragma unroll
            for (int r = 0; r < 8; ++r)
                oacc[c][r] *= alpha[r];

        // ---- stage P (16x32) through LDS to rebuild it in A-layout -------
        #pragma unroll
        for (int r = 0; r < 8; ++r) {
            const int row = r + 8 * h;
            lp[row * 32 + ln]      = (__bf16)s0[r];
            lp[row * 32 + 16 + ln] = (__bf16)s1[r];
        }
        asm volatile("s_wait_dscnt 0" ::: "memory");
        __builtin_amdgcn_wave_barrier();

        const __bf16* pr = lp + ln * 32 + h * 8;
        v16bf pf = frag_bf16(pr, pr + 16);

        asm volatile("s_wait_dscnt 0" ::: "memory");
        __builtin_amdgcn_wave_barrier();

        // ---- l += P * 1 ;  O += P * V  (V bf16 (C,P): pure b128 loads) ---
        lacc = __builtin_amdgcn_wmma_f32_16x16x32_bf16(
                   false, pf, false, ones, (short)0, lacc, false, false);
        const __bf16* vbase = v_cp + (size_t)b * CO * PIX + cbase + j0;
        #pragma unroll
        for (int c = 0; c < 8; ++c) {
            const __bf16* vp = vbase + (size_t)(c * 16 + ln) * PIX + h * 8;
            v16bf vf = frag_bf16(vp, vp + 16);
            oacc[c] = __builtin_amdgcn_wmma_f32_16x16x32_bf16(
                          false, pf, false, vf, (short)0, oacc[c], false, false);
        }
    }

    // ---- modality softmax + blend + store --------------------------------
    const float a = mw[0], bb = mw[1];
    const float mx = fmaxf(a, bb);
    const float e0 = __expf(a - mx), e1 = __expf(bb - mx);
    const float w0 = e0 / (e0 + e1), w1 = e1 / (e0 + e1);

    float rinv[8];
    #pragma unroll
    for (int r = 0; r < 8; ++r) rinv[r] = 1.0f / lacc[r];

    #pragma unroll
    for (int c = 0; c < 8; ++c) {
        #pragma unroll
        for (int r = 0; r < 8; ++r) {
            const int    cc  = c * 16 + ln;
            const size_t p   = cbase + q0 + r + 8 * h;
            const size_t idx = ((size_t)b * CO + cc) * PIX + p;
            const float fused = oacc[c][r] * rinv[r];
            out[idx] = w0 * i_cp[idx] + w1 * fused;
        }
    }
}

// ---------------------------------------------------------------------------
extern "C" void kernel_launch(void* const* d_in, const int* in_sizes, int n_in,
                              void* d_out, int out_size, void* d_ws, size_t ws_size,
                              hipStream_t stream)
{
    (void)in_sizes; (void)n_in; (void)out_size; (void)ws_size;

    const float* lidar_x  = (const float*)d_in[0];
    const float* image_x  = (const float*)d_in[1];
    const float* lidar_w  = (const float*)d_in[2];
    const float* lidar_g  = (const float*)d_in[3];
    const float* lidar_b  = (const float*)d_in[4];
    const float* lidar_m  = (const float*)d_in[5];
    const float* lidar_v  = (const float*)d_in[6];
    const float* image_w  = (const float*)d_in[7];
    const float* image_g  = (const float*)d_in[8];
    const float* image_b  = (const float*)d_in[9];
    const float* image_m  = (const float*)d_in[10];
    const float* image_v  = (const float*)d_in[11];
    const float* mweights = (const float*)d_in[12];

    char* ws = (char*)d_ws;
    float*  imageProj = (float*) (ws);                         // 16 MB f32 (C,P)
    __bf16* lidarV    = (__bf16*)(ws + (size_t)16777216);      //  8 MB bf16 (C,P)
    __bf16* lidarT    = (__bf16*)(ws + (size_t)25165824);      //  8 MB bf16 (P,C)
    __bf16* imageT    = (__bf16*)(ws + (size_t)33554432);      //  8 MB bf16 (P,C)

    const float qk_scale = 0.08838834764831845f;  // 1/sqrt(128), folded into Q

    // projections: 16384 waves each -> 4096 blocks of 128 threads
    proj_kernel<false><<<4096, 128, 0, stream>>>(
        lidar_x, lidar_w, lidar_g, lidar_b, lidar_m, lidar_v,
        (float*)nullptr, lidarV, lidarT, 256, 1.0f);
    proj_kernel<true><<<4096, 128, 0, stream>>>(
        image_x, image_w, image_g, image_b, image_m, image_v,
        imageProj, (__bf16*)nullptr, imageT, 512, qk_scale);

    // attention + blend: 2048 waves -> 512 blocks of 128 threads
    attn_kernel<<<512, 128, 0, stream>>>(imageT, lidarT, lidarV, imageProj,
                                         mweights, (float*)d_out);
}
